// nnue_43748536877078
// MI455X (gfx1250) — compile-verified, matrix operand order checked by reading, not run on hardware
//
#include <hip/hip_runtime.h>

typedef _Float16 v16h  __attribute__((ext_vector_type(16)));
typedef _Float16 half8 __attribute__((ext_vector_type(8)));
typedef float    v8f   __attribute__((ext_vector_type(8)));
typedef float    f4    __attribute__((ext_vector_type(4)));

#define FT_IN  23232
#define NCOL   520      // K_HALF + BUCKETS
#define N_PAD  576      // padded weight rows (multiple of N_BLK, zero-filled)
#define KHALF  512
#define BROWS  4096
#define M_BLK  128
#define N_BLK  64
#define K_BLK  64
#define LDA    72       // padded LDS stride in halves (144 B, keeps b128 alignment)
#define LDB    72

union Frag { v16h v; half8 h[2]; };

// ---------------------------------------------------------------------------
// Pass 0: pre-convert ft_w (520 x 23232 f32) -> f16, padded to 576 rows of 0.
// Read once (48 MB), written once (27 MB); saves 64x redundant conversion.
// ---------------------------------------------------------------------------
__global__ __launch_bounds__(256)
void cvt_w16(const float* __restrict__ w, _Float16* __restrict__ o)
{
    const int n = blockIdx.y;
    const int k = (blockIdx.x * 256 + threadIdx.x) * 8;
    if (k >= FT_IN) return;
    half8 h;
    if (n < NCOL) {
        f4 x0 = *(const f4*)(w + (long)n * FT_IN + k);
        f4 x1 = *(const f4*)(w + (long)n * FT_IN + k + 4);
        #pragma unroll
        for (int i = 0; i < 4; ++i) { h[i] = (_Float16)x0[i]; h[i + 4] = (_Float16)x1[i]; }
    } else {
        #pragma unroll
        for (int i = 0; i < 8; ++i) h[i] = (_Float16)0.0f;
    }
    *(half8*)(o + (long)n * FT_IN + k) = h;
}

// ---------------------------------------------------------------------------
// Stage 1: fused feature-transform GEMM (virtual M = 8192 over f1|f2).
//   A: f32 features, converted f32->f16 in registers, staged to LDS.
//   B: pre-converted f16 weights, copied HBM->LDS with async-to-LDS (no VGPRs).
//   8 x V_WMMA_F32_16X16X32_F16 per wave per K_BLK=64 iteration.
// ---------------------------------------------------------------------------
__global__ __launch_bounds__(256)
void ft_gemm_wmma(const float* __restrict__ f1, const float* __restrict__ f2,
                  const _Float16* __restrict__ w16, const float* __restrict__ bias,
                  float* __restrict__ out)
{
    __shared__ _Float16 sA[M_BLK * LDA];
    __shared__ _Float16 sB[N_BLK * LDB];

    const int tid  = threadIdx.x;
    const int lane = tid & 31;
    const int wave = tid >> 5;
    const int wm   = wave >> 1;          // 0..3  (M sub-tile)
    const int wn   = wave & 1;           // 0..1  (N sub-tile)
    const int mh   = lane & 15;
    const int kh   = lane >> 4;

    const int mTile = blockIdx.x * M_BLK;        // virtual row base 0..8191
    const int nTile = blockIdx.y * N_BLK;

    const float* A = (mTile < BROWS) ? f1 : f2;
    const long aRowBase = (long)((mTile < BROWS) ? mTile : mTile - BROWS);

    // A staging: 2 threads per row, 32 f32 each (k segment sseg*32)
    const int  sm   = tid >> 1;
    const int  sseg = tid & 1;
    const long aOff = (aRowBase + sm) * (long)FT_IN + sseg * 32;
    _Float16*  aDst = &sA[sm * LDA + sseg * 32];

    // B staging: 4 threads per weight row, 16 halves each, async HBM->LDS.
    const int bn   = tid >> 2;
    const int bseg = tid & 3;
    const _Float16* bSrcBase = w16 + (long)(nTile + bn) * FT_IN + bseg * 16;
    const unsigned  bDst = (unsigned)(unsigned long long)&sB[bn * LDB + bseg * 16];

    v8f acc[2][2] = {};

    for (int kt = 0; kt < FT_IN; kt += K_BLK) {
        // ---- B tile: two async b128 copies (offset applies to LDS + global) ----
        const _Float16* bp = bSrcBase + kt;
        asm volatile(
            "global_load_async_to_lds_b128 %0, %1, off\n\t"
            "global_load_async_to_lds_b128 %0, %1, off offset:16"
            :: "v"(bDst), "v"(bp) : "memory");

        // ---- A tile: load f32, convert to f16, store to LDS ----
        f4 a0 = *(const f4*)(A + aOff + kt);
        f4 a1 = *(const f4*)(A + aOff + kt + 4);
        f4 a2 = *(const f4*)(A + aOff + kt + 8);
        f4 a3 = *(const f4*)(A + aOff + kt + 12);
        f4 a4 = *(const f4*)(A + aOff + kt + 16);
        f4 a5 = *(const f4*)(A + aOff + kt + 20);
        f4 a6 = *(const f4*)(A + aOff + kt + 24);
        f4 a7 = *(const f4*)(A + aOff + kt + 28);
        half8 h0, h1, h2, h3;
        #pragma unroll
        for (int i = 0; i < 4; ++i) {
            h0[i] = (_Float16)a0[i]; h0[i + 4] = (_Float16)a1[i];
            h1[i] = (_Float16)a2[i]; h1[i + 4] = (_Float16)a3[i];
            h2[i] = (_Float16)a4[i]; h2[i + 4] = (_Float16)a5[i];
            h3[i] = (_Float16)a6[i]; h3[i + 4] = (_Float16)a7[i];
        }
        *(half8*)(aDst)      = h0;
        *(half8*)(aDst + 8)  = h1;
        *(half8*)(aDst + 16) = h2;
        *(half8*)(aDst + 24) = h3;

        if (kt + K_BLK < FT_IN)
            __builtin_prefetch(A + aOff + kt + K_BLK, 0, 1);   // global_prefetch_b8

        asm volatile("s_wait_asynccnt 0" ::: "memory");        // B tile landed in LDS
        __syncthreads();

        // ---- 2 K-substeps of 32: fragment loads + 4 WMMAs each ----
        #pragma unroll
        for (int ks = 0; ks < 2; ++ks) {
            Frag fa[2], fb[2];
            #pragma unroll
            for (int i = 0; i < 2; ++i) {
                const _Float16* pa = &sA[(wm * 32 + i * 16 + mh) * LDA + ks * 32 + kh * 8];
                fa[i].h[0] = *(const half8*)pa;          // K = 8*kh .. +7
                fa[i].h[1] = *(const half8*)(pa + 16);   // K = 16 + 8*kh .. +7
                const _Float16* pb = &sB[(wn * 32 + i * 16 + mh) * LDB + ks * 32 + kh * 8];
                fb[i].h[0] = *(const half8*)pb;
                fb[i].h[1] = *(const half8*)(pb + 16);
            }
            #pragma unroll
            for (int i = 0; i < 2; ++i)
                #pragma unroll
                for (int j = 0; j < 2; ++j)
                    acc[i][j] = __builtin_amdgcn_wmma_f32_16x16x32_f16(
                        false, fa[i].v, false, fb[j].v,
                        (short)0, acc[i][j], false, false);
        }
        __syncthreads();
    }

    // ---- epilogue: add ft_b, store f32 ----
    #pragma unroll
    for (int j = 0; j < 2; ++j) {
        const int col = nTile + wn * 32 + j * 16 + mh;
        if (col >= NCOL) continue;
        const float bv = bias[col];
        #pragma unroll
        for (int i = 0; i < 2; ++i) {
            const int rowBase = mTile + wm * 32 + i * 16 + kh * 8;
            #pragma unroll
            for (int r = 0; r < 8; ++r)
                out[(long)(rowBase + r) * NCOL + col] = acc[i][j][r] + bv;
        }
    }
}

// ---------------------------------------------------------------------------
// Stage 2: per-row bucketed tail. One wave32 per batch row.
// ---------------------------------------------------------------------------
__global__ __launch_bounds__(256)
void nnue_tail(const float* __restrict__ acc, const int* __restrict__ bucket,
               const float* __restrict__ h1_w, const float* __restrict__ h1_b,
               const float* __restrict__ h2_w, const float* __restrict__ h2_b,
               const float* __restrict__ out_w, const float* __restrict__ out_b,
               float* __restrict__ out)
{
    __shared__ float sh1[8][16];

    const int lane = threadIdx.x & 31;
    const int wv   = threadIdx.x >> 5;
    const int row  = blockIdx.x * 8 + wv;

    const int b = bucket[row];
    const float* a1 = acc + (long)row * NCOL;
    const float* a2 = acc + (long)(BROWS + row) * NCOL;
    const float psqt = 0.5f * (a1[KHALF + b] - a2[KHALF + b]);

    const int j    = lane & 15;
    const int hsel = lane >> 4;
    const float* src = hsel ? a2 : a1;
    const float* w1  = h1_w + (long)(b * 16 + j) * (2 * KHALF) + hsel * KHALF;

    float s = 0.f;
    for (int k = 0; k < KHALF; k += 4) {
        f4 v  = *(const f4*)(src + k);
        f4 wr = *(const f4*)(w1 + k);
        #pragma unroll
        for (int e = 0; e < 4; ++e) {
            float c = fminf(fmaxf(v[e], 0.f), 1.f);
            s += c * wr[e];
        }
    }
    s += __shfl_xor(s, 16, 32);
    if (hsel == 0) {
        float h1v = s + h1_b[b * 16 + j];
        sh1[wv][j] = fminf(fmaxf(h1v, 0.f), 1.f);
    }
    __syncthreads();

    const float* w2 = h2_w + (b * 32 + lane) * 16;
    float s2 = 0.f;
    #pragma unroll
    for (int k = 0; k < 16; ++k) s2 += sh1[wv][k] * w2[k];
    const float h2v = fminf(fmaxf(s2 + h2_b[b * 32 + lane], 0.f), 1.f);

    float term = h2v * out_w[b * 32 + lane];
    #pragma unroll
    for (int off = 16; off; off >>= 1) term += __shfl_xor(term, off, 32);

    if (lane == 0) out[row] = term + out_b[b] + 0.25f * psqt;
}

extern "C" void kernel_launch(void* const* d_in, const int* in_sizes, int n_in,
                              void* d_out, int out_size, void* d_ws, size_t ws_size,
                              hipStream_t stream)
{
    (void)in_sizes; (void)n_in; (void)out_size; (void)ws_size;
    const int*   bucket = (const int*)  d_in[0];
    const float* f1     = (const float*)d_in[1];
    const float* f2     = (const float*)d_in[2];
    const float* ft_w   = (const float*)d_in[3];
    const float* ft_b   = (const float*)d_in[4];
    const float* h1_w   = (const float*)d_in[5];
    const float* h1_b   = (const float*)d_in[6];
    const float* h2_w   = (const float*)d_in[7];
    const float* h2_b   = (const float*)d_in[8];
    const float* out_w  = (const float*)d_in[9];
    const float* out_b  = (const float*)d_in[10];
    float* out = (float*)d_out;

    // workspace: [ acc: 8192*520 f32 = 17,039,360 B ][ w16: 576*23232 f16 ]
    float*    accbuf = (float*)d_ws;
    _Float16* w16    = (_Float16*)((char*)d_ws + (size_t)(2 * BROWS) * NCOL * sizeof(float));

    dim3 gc((FT_IN / 8 + 255) / 256, N_PAD);                  // 12 x 576
    cvt_w16<<<gc, 256, 0, stream>>>(ft_w, w16);

    dim3 g1(2 * BROWS / M_BLK, N_PAD / N_BLK);                // 64 x 9
    ft_gemm_wmma<<<g1, 256, 0, stream>>>(f1, f2, w16, ft_b, accbuf);

    nnue_tail<<<BROWS / 8, 256, 0, stream>>>(accbuf, bucket,
                                             h1_w, h1_b, h2_w, h2_b,
                                             out_w, out_b, out);
}
